// PAM_17978733101351
// MI455X (gfx1250) — matrix operand order repaired and checked by reading, not compiled, not run on hardware
//
#include <hip/hip_runtime.h>
#include <stdint.h>

#define H   256
#define W   512
#define HWT (H*W)        // 131072
#define C4  256
#define CCH 64

typedef __bf16 bf16_t;
typedef bf16_t v16bf __attribute__((ext_vector_type(16)));
typedef float  v8f   __attribute__((ext_vector_type(8)));

__device__ __forceinline__ uint16_t f2bf(float f) {
    uint32_t u = __float_as_uint(f);
    u += 0x7fffu + ((u >> 16) & 1u);      // round-to-nearest-even
    return (uint16_t)(u >> 16);
}
__device__ __forceinline__ float bf2f(uint16_t h) {
    return __uint_as_float(((uint32_t)h) << 16);
}
__device__ __forceinline__ v16bf mk_frag(uint4 a, uint4 b) {
    union { uint4 q[2]; v16bf v; } u;
    u.q[0] = a; u.q[1] = b;
    return u.v;
}
__device__ __forceinline__ uint4 pack8bf(const float* v) {
    uint4 r;
    r.x = (uint32_t)f2bf(v[0]) | ((uint32_t)f2bf(v[1]) << 16);
    r.y = (uint32_t)f2bf(v[2]) | ((uint32_t)f2bf(v[3]) << 16);
    r.z = (uint32_t)f2bf(v[4]) | ((uint32_t)f2bf(v[5]) << 16);
    r.w = (uint32_t)f2bf(v[6]) | ((uint32_t)f2bf(v[7]) << 16);
    return r;
}

// ---------------------------------------------------------------------------
// Weight conversion: rb_w (oc,ic,3,3) f32 -> (oc, tap, ic) bf16 ; qk (64,64)
// ---------------------------------------------------------------------------
__global__ void wcvt_kernel(const float* __restrict__ w1, const float* __restrict__ w2,
                            const float* __restrict__ qw, const float* __restrict__ sw,
                            uint16_t* __restrict__ w1b, uint16_t* __restrict__ w2b,
                            uint16_t* __restrict__ qwb, uint16_t* __restrict__ swb) {
    int i = blockIdx.x * 256 + threadIdx.x;
    if (i < 256 * 9 * 64) {
        int oc = i / 576, r = i % 576, tap = r / 64, ic = r % 64;
        int src = oc * 576 + ic * 9 + tap;
        w1b[i] = f2bf(w1[src]);
        w2b[i] = f2bf(w2[src]);
    }
    if (i < 64 * 64) { qwb[i] = f2bf(qw[i]); swb[i] = f2bf(sw[i]); }
}

// ---------------------------------------------------------------------------
// BN (eval) + NCHW->NHWC transpose, f32 -> bf16.  tile 64ch x 64px via LDS.
// ---------------------------------------------------------------------------
__global__ __launch_bounds__(256) void bn_kernel(
    const float* __restrict__ x, const float* __restrict__ gam,
    const float* __restrict__ bet, const float* __restrict__ mu,
    const float* __restrict__ var, uint16_t* __restrict__ out) {
    __shared__ float tile[64][65];
    const int t = threadIdx.x;
    const int p0 = blockIdx.x * 64;
    const int c0 = blockIdx.y * 64;
    {
        int px = t & 63, cg = t >> 6;
        for (int k = 0; k < 16; ++k) {
            int ch = cg * 16 + k;
            int c = c0 + ch;
            float inv = gam[c] * rsqrtf(var[c] + 1e-5f);
            float b2 = bet[c] - mu[c] * inv;
            tile[ch][px] = x[(size_t)c * HWT + p0 + px] * inv + b2;
        }
    }
    __syncthreads();
    {
        int ch = t & 63, pg = t >> 6;
        for (int k = 0; k < 16; ++k) {
            int px = pg * 16 + k;
            out[(size_t)(p0 + px) * C4 + c0 + ch] = f2bf(tile[ch][px]);
        }
    }
}

// ---------------------------------------------------------------------------
// Grouped 3x3 conv (groups=4, 64ic/64oc per group), pad 1, implicit GEMM.
// in/out: (HW, 256) bf16 channel-last.  mode 0: LeakyReLU(0.1); 1: +residual.
// Block: 128 thr (4 waves), tile = 64 oc x 64 px (row y, cols x0..x0+63).
// Input tile staged with CDNA5 async global->LDS copies (ASYNCcnt).
// ---------------------------------------------------------------------------
__global__ __launch_bounds__(128) void conv3x3_kernel(
    const uint16_t* __restrict__ in, const uint16_t* __restrict__ wts,
    const float* __restrict__ bias, const uint16_t* __restrict__ resid,
    uint16_t* __restrict__ out, int mode) {
    __shared__ __align__(16) uint16_t smem[3][66][72];
    const int t = threadIdx.x;
    const int x0 = blockIdx.x * 64;
    const int y = blockIdx.y;
    const int g = blockIdx.z;

    // pull this group's weight block toward the caches while we stage input
    __builtin_prefetch(wts + (size_t)(g * 64 + (t & 63)) * 576, 0, 3);

    for (int q = t; q < 3 * 66 * 8; q += 128) {      // stage 3 rows x 66 cols x 64 ic
        int oct = q & 7;
        int rc = q >> 3;
        int col = rc % 66;
        int row = rc / 66;
        int gy = y + row - 1;
        int gx = x0 + col - 1;
        uint16_t* dstp = &smem[row][col][oct * 8];
        if ((unsigned)gy < (unsigned)H && (unsigned)gx < (unsigned)W) {
            const uint16_t* srcp = in + ((size_t)gy * W + gx) * C4 + g * 64 + oct * 8;
            uint32_t lds = (uint32_t)(uintptr_t)dstp;    // low 32b = LDS byte address
            asm volatile("global_load_async_to_lds_b128 %0, %1, off"
                         :: "v"(lds), "v"(srcp) : "memory");
        } else {
            *(uint4*)dstp = make_uint4(0u, 0u, 0u, 0u);  // zero halo
        }
    }
    asm volatile("s_wait_asynccnt 0" ::: "memory");       // drain async LDS writes
    __syncthreads();

    const int l = t & 31, wv = t >> 5;
    const int ln = l & 15, half = l >> 4;
    const int mt = wv;                         // 16-oc tile per wave
    v8f acc[4] = {};
    for (int kc = 0; kc < 18; ++kc) {          // K = 9 taps * 64 ic, chunks of 32
        int tap = kc >> 1, ic0 = (kc & 1) * 32;
        int dy = tap / 3, dx = tap % 3;
        const uint16_t* ap = wts + ((size_t)(g * 64 + mt * 16 + ln) * 9 + tap) * 64 + ic0 + half * 8;
        v16bf a = mk_frag(*(const uint4*)ap, *(const uint4*)(ap + 16));
#pragma unroll
        for (int nt = 0; nt < 4; ++nt) {
            const uint16_t* bp = &smem[dy][nt * 16 + ln + dx][ic0 + half * 8];
            v16bf b = mk_frag(*(const uint4*)bp, *(const uint4*)(bp + 16));
            acc[nt] = __builtin_amdgcn_wmma_f32_16x16x32_bf16(
                false, a, false, b, (short)0, acc[nt], false, false);
        }
    }
    const int chbase = g * 64 + mt * 16 + half * 8;
#pragma unroll
    for (int nt = 0; nt < 4; ++nt) {
        int x = x0 + nt * 16 + ln;
        size_t off = ((size_t)y * W + x) * C4 + chbase;
        float vals[8];
#pragma unroll
        for (int r = 0; r < 8; ++r) vals[r] = acc[nt][r] + bias[chbase + r];
        if (mode == 0) {
#pragma unroll
            for (int r = 0; r < 8; ++r) vals[r] = vals[r] >= 0.f ? vals[r] : 0.1f * vals[r];
        } else {
            uint4 rv = *(const uint4*)(resid + off);
            const uint16_t* rp = (const uint16_t*)&rv;
#pragma unroll
            for (int r = 0; r < 8; ++r) vals[r] += bf2f(rp[r]);
        }
        *(uint4*)(out + off) = pack8bf(vals);
    }
}

// ---------------------------------------------------------------------------
// Grouped 1x1 conv (groups=4: 16oc x 64ic per group), feat bf16 -> Q/K f32.
// Block 256 (8 waves): wave -> (group, 64px half), 128 px per block.
// ---------------------------------------------------------------------------
__global__ __launch_bounds__(256) void conv1x1_kernel(
    const uint16_t* __restrict__ feat, const uint16_t* __restrict__ wq,
    const float* __restrict__ bias, float* __restrict__ outQ) {
    const int t = threadIdx.x;
    const int l = t & 31, wv = t >> 5;
    const int ln = l & 15, half = l >> 4;
    const int g = wv & 3;
    const int pb = blockIdx.x * 128 + (wv >> 2) * 64;
    v8f acc[4] = {};
    for (int kc = 0; kc < 2; ++kc) {
        const uint16_t* ap = wq + (size_t)(g * 16 + ln) * 64 + kc * 32 + half * 8;
        v16bf a = mk_frag(*(const uint4*)ap, *(const uint4*)(ap + 16));
#pragma unroll
        for (int nt = 0; nt < 4; ++nt) {
            int pix = pb + nt * 16 + ln;
            const uint16_t* bp = feat + (size_t)pix * C4 + g * 64 + kc * 32 + half * 8;
            v16bf b = mk_frag(*(const uint4*)bp, *(const uint4*)(bp + 16));
            acc[nt] = __builtin_amdgcn_wmma_f32_16x16x32_bf16(
                false, a, false, b, (short)0, acc[nt], false, false);
        }
    }
    const int chb = g * 16 + half * 8;
#pragma unroll
    for (int nt = 0; nt < 4; ++nt) {
        int pix = pb + nt * 16 + ln;
        float4 o0 = make_float4(acc[nt][0] + bias[chb + 0], acc[nt][1] + bias[chb + 1],
                                acc[nt][2] + bias[chb + 2], acc[nt][3] + bias[chb + 3]);
        float4 o1 = make_float4(acc[nt][4] + bias[chb + 4], acc[nt][5] + bias[chb + 5],
                                acc[nt][6] + bias[chb + 6], acc[nt][7] + bias[chb + 7]);
        *(float4*)(outQ + (size_t)pix * 64 + chb) = o0;
        *(float4*)(outQ + (size_t)pix * 64 + chb + 4) = o1;
    }
}

// ---------------------------------------------------------------------------
// Disparity gathers -> flat (h,w,c) buffers (faithful to the torch reshape:
// these are later *read* as (c,h,w)).
// ---------------------------------------------------------------------------
__global__ __launch_bounds__(256) void gather_kernel(
    const float* __restrict__ Qf, const float* __restrict__ Kf,
    const float* __restrict__ xl, const float* __restrict__ xr,
    const float* __restrict__ dl, const float* __restrict__ dr,
    float* __restrict__ Gq, float* __restrict__ Gk,
    float* __restrict__ Gxl, float* __restrict__ Gxr) {
    int gt = blockIdx.x * 256 + threadIdx.x;  // < HW*64
    int pix = gt >> 6, cc = gt & 63;
    int ww = pix & (W - 1), hh = pix >> 9;
    float dlv = dl[pix], drv = dr[pix];
    int wr = (int)fmaxf((float)ww + 0.5f - dlv, 0.f);            // r2l
    int wl = (int)fminf((float)ww + 0.5f + drv, (float)(W - 1)); // l2r
    int rowb = hh * W;
    Gk[gt]  = Kf[(size_t)(rowb + wr) * 64 + cc];
    Gq[gt]  = Qf[(size_t)(rowb + wl) * 64 + cc];
    Gxr[gt] = xr[(size_t)cc * HWT + rowb + wr];
    Gxl[gt] = xl[(size_t)cc * HWT + rowb + wl];
}

// ---------------------------------------------------------------------------
// 64x64x64 bf16 WMMA matmul helper: D[m][n] = sum_k A[m][k]*B_buf[n][k].
// A,B row-major bf16 [64][64] in LDS; O f32 with row stride `os`.
// ---------------------------------------------------------------------------
__device__ __forceinline__ void mm64_bf16(const uint16_t* Abuf, const uint16_t* Bbuf,
                                          float* Obuf, int os, int tid) {
    const int l = tid & 31, wv = tid >> 5;
    const int ln = l & 15, half = l >> 4;
    for (int tile = wv; tile < 16; tile += 8) {
        int mt = tile >> 2, nt = tile & 3;
        v8f acc = {};
#pragma unroll
        for (int kc = 0; kc < 2; ++kc) {
            const uint16_t* ap = Abuf + (mt * 16 + ln) * 64 + kc * 32 + half * 8;
            v16bf a = mk_frag(*(const uint4*)ap, *(const uint4*)(ap + 16));
            const uint16_t* bp = Bbuf + (nt * 16 + ln) * 64 + kc * 32 + half * 8;
            v16bf b = mk_frag(*(const uint4*)bp, *(const uint4*)(bp + 16));
            acc = __builtin_amdgcn_wmma_f32_16x16x32_bf16(
                false, a, false, b, (short)0, acc, false, false);
        }
#pragma unroll
        for (int r = 0; r < 8; ++r)
            Obuf[(mt * 16 + half * 8 + r) * os + nt * 16 + ln] = acc[r];
    }
}

// ---------------------------------------------------------------------------
// Per-window PAM: scores -> softmax -> relax/V/tanh -> M@x_sel -> blend.
// One block per 8x8 window (grid 64x32), 256 threads (8 waves).
// ---------------------------------------------------------------------------
__global__ __launch_bounds__(256) void pam_kernel(
    const float* __restrict__ Qf, const float* __restrict__ Kf,
    const float* __restrict__ Gq, const float* __restrict__ Gk,
    const float* __restrict__ Gxl, const float* __restrict__ Gxr,
    const float* __restrict__ xl, const float* __restrict__ xr,
    float* __restrict__ outL, float* __restrict__ outR) {
    __shared__ float sMr[64][65];
    __shared__ float sMl[64][65];
    __shared__ __align__(16) uint16_t sA[64 * 64];
    __shared__ __align__(16) uint16_t sA2[64 * 64];
    __shared__ __align__(16) uint16_t sB[64 * 64];
    __shared__ float sMean[64];
    __shared__ float sV[128];

    const int t = threadIdx.x;
    const int x0 = blockIdx.x * 8;
    const int y0 = blockIdx.y * 8;
    const int pixbase = y0 * W + x0;

    // --- P1: Q window means; Bk <- Gk (viewed as (c,h,w)) ---
    if (t < 64) {
        float s = 0.f;
        for (int p = 0; p < 64; ++p)
            s += Qf[(size_t)(pixbase + (p >> 3) * W + (p & 7)) * 64 + t];
        sMean[t] = s * (1.f / 64.f);
    }
    for (int task = t; task < 512; task += 256) {
        int cc = task >> 3, i = task & 7;
        const float* src = Gk + (size_t)cc * HWT + pixbase + i * W;
        for (int j = 0; j < 8; ++j) sB[(i * 8 + j) * 64 + cc] = f2bf(src[j]);
    }
    __syncthreads();
    // --- P2: centered Q -> sA (bf16) ---
    {
        int p = t >> 2, c0 = (t & 3) * 16;
        const float* src = Qf + (size_t)(pixbase + (p >> 3) * W + (p & 7)) * 64;
        for (int k = 0; k < 16; ++k)
            sA[p * 64 + c0 + k] = f2bf(src[c0 + k] - sMean[c0 + k]);
    }
    __syncthreads();
    mm64_bf16(sA, sB, &sMr[0][0], 65, t);   // score_r2l
    __syncthreads();
    // --- P4: K means; Bq <- Gq ---
    if (t < 64) {
        float s = 0.f;
        for (int p = 0; p < 64; ++p)
            s += Kf[(size_t)(pixbase + (p >> 3) * W + (p & 7)) * 64 + t];
        sMean[t] = s * (1.f / 64.f);
    }
    for (int task = t; task < 512; task += 256) {
        int cc = task >> 3, i = task & 7;
        const float* src = Gq + (size_t)cc * HWT + pixbase + i * W;
        for (int j = 0; j < 8; ++j) sB[(i * 8 + j) * 64 + cc] = f2bf(src[j]);
    }
    __syncthreads();
    {
        int p = t >> 2, c0 = (t & 3) * 16;
        const float* src = Kf + (size_t)(pixbase + (p >> 3) * W + (p & 7)) * 64;
        for (int k = 0; k < 16; ++k)
            sA[p * 64 + c0 + k] = f2bf(src[c0 + k] - sMean[c0 + k]);
    }
    __syncthreads();
    mm64_bf16(sA, sB, &sMl[0][0], 65, t);   // score_l2r
    __syncthreads();
    // --- softmax rows (f32) ---
    if (t < 128) {
        float* row = (t < 64) ? &sMr[t][0] : &sMl[t - 64][0];
        float m = row[0];
        for (int j = 1; j < 64; ++j) m = fmaxf(m, row[j]);
        float s = 0.f;
        for (int j = 0; j < 64; ++j) { float e = __expf(row[j] - m); row[j] = e; s += e; }
        float inv = 1.f / s;
        for (int j = 0; j < 64; ++j) row[j] *= inv;
    }
    __syncthreads();
    // --- relax + V + tanh ; Amr->sA, Aml->sA2 ; Xr_b -> sB ---
    {
        int p = t >> 2, c0 = (t & 3) * 16;
        for (int k = 0; k < 16; ++k) {
            sA[p * 64 + c0 + k]  = f2bf(sMr[p][c0 + k]);
            sA2[p * 64 + c0 + k] = f2bf(sMl[p][c0 + k]);
        }
    }
    if (t < 128) {
        int p = t & 63;
        bool left = (t < 64);
        int klo = p - 2 < 0 ? 0 : p - 2;
        int khi = p + 2 > 63 ? 63 : p + 2;
        float acc = 0.f;
        for (int j = 0; j < 64; ++j) {
            float s = 0.f;
            if (left) { for (int k = klo; k <= khi; ++k) s += sMr[k][j]; acc += s * sMl[j][p]; }
            else      { for (int k = klo; k <= khi; ++k) s += sMl[k][j]; acc += s * sMr[j][p]; }
        }
        sV[t] = tanhf(5.f * acc);
    }
    for (int task = t; task < 512; task += 256) {
        int cc = task >> 3, i = task & 7;
        const float* src = Gxr + (size_t)cc * HWT + pixbase + i * W;
        float v[8];
        for (int j = 0; j < 8; ++j) v[j] = src[j];
        *(uint4*)&sB[cc * 64 + i * 8] = pack8bf(v);   // layout [cc][p2]
    }
    __syncthreads();
    mm64_bf16(sA, sB, &sMl[0][0], 65, t);   // XT_left = Mr2l @ x_r_sel (reuse sMl)
    __syncthreads();
    // --- blend left ; load Xl_b -> sB ---
    for (int task = t; task < 512; task += 256) {
        int cc = task >> 3, i = task & 7;
        const float* xs = xl + (size_t)cc * HWT + pixbase + i * W;
        float* dst = outL + (size_t)cc * HWT + pixbase + i * W;
        for (int j = 0; j < 8; ++j) {
            int p = i * 8 + j;
            float V = sV[p];
            dst[j] = xs[j] * (1.f - V) + (&sMl[0][0])[p * 65 + cc] * V;
        }
    }
    for (int task = t; task < 512; task += 256) {
        int cc = task >> 3, i = task & 7;
        const float* src = Gxl + (size_t)cc * HWT + pixbase + i * W;
        float v[8];
        for (int j = 0; j < 8; ++j) v[j] = src[j];
        *(uint4*)&sB[cc * 64 + i * 8] = pack8bf(v);
    }
    __syncthreads();
    mm64_bf16(sA2, sB, &sMr[0][0], 65, t);  // XT_right = Ml2r @ x_l_sel (reuse sMr)
    __syncthreads();
    for (int task = t; task < 512; task += 256) {
        int cc = task >> 3, i = task & 7;
        const float* xs = xr + (size_t)cc * HWT + pixbase + i * W;
        float* dst = outR + (size_t)cc * HWT + pixbase + i * W;
        for (int j = 0; j < 8; ++j) {
            int p = i * 8 + j;
            float V = sV[64 + p];
            dst[j] = xs[j] * (1.f - V) + (&sMr[0][0])[p * 65 + cc] * V;
        }
    }
}

// ---------------------------------------------------------------------------
extern "C" void kernel_launch(void* const* d_in, const int* in_sizes, int n_in,
                              void* d_out, int out_size, void* d_ws, size_t ws_size,
                              hipStream_t stream) {
    (void)in_sizes; (void)n_in; (void)out_size; (void)ws_size;
    const float* x_left  = (const float*)d_in[0];
    const float* x_right = (const float*)d_in[1];
    const float* cat_l   = (const float*)d_in[2];
    const float* cat_r   = (const float*)d_in[3];
    const float* d_left  = (const float*)d_in[4];
    const float* d_right = (const float*)d_in[5];
    const float* gam = (const float*)d_in[6];
    const float* bet = (const float*)d_in[7];
    const float* mu  = (const float*)d_in[8];
    const float* var = (const float*)d_in[9];
    const float* w1  = (const float*)d_in[10];
    const float* b1  = (const float*)d_in[11];
    const float* w2  = (const float*)d_in[12];
    const float* b2  = (const float*)d_in[13];
    const float* qw  = (const float*)d_in[14];
    const float* qb  = (const float*)d_in[15];
    const float* sw  = (const float*)d_in[16];
    const float* sb  = (const float*)d_in[17];

    const size_t BN_BYTES = (size_t)HWT * C4 * 2;   // 64 MiB
    const size_t QF_BYTES = (size_t)HWT * 64 * 4;   // 32 MiB
    uint8_t* ws = (uint8_t*)d_ws;
    uint16_t* bnbuf = (uint16_t*)(ws);
    uint16_t* ybuf  = (uint16_t*)(ws + BN_BYTES);
    uint16_t* fbuf  = (uint16_t*)(ws + 2 * BN_BYTES);
    float*    Qfb   = (float*)(ws + 3 * BN_BYTES);
    float*    Kfb   = (float*)(ws + 3 * BN_BYTES + QF_BYTES);
    uint16_t* w1b   = (uint16_t*)(ws + 3 * BN_BYTES + 2 * QF_BYTES);
    uint16_t* w2b   = w1b + 256 * 9 * 64;
    uint16_t* qwb   = w2b + 256 * 9 * 64;
    uint16_t* swb   = qwb + 64 * 64;
    // gather outputs overlay bnbuf/ybuf (dead by then)
    float* Gq  = (float*)(ws);
    float* Gk  = (float*)(ws + QF_BYTES);
    float* Gxl = (float*)(ws + BN_BYTES);
    float* Gxr = (float*)(ws + BN_BYTES + QF_BYTES);

    float* outL = (float*)d_out;
    float* outR = outL + (size_t)CCH * HWT;

    wcvt_kernel<<<576, 256, 0, stream>>>(w1, w2, qw, sw, w1b, w2b, qwb, swb);

    // ---- left branch ----
    bn_kernel<<<dim3(HWT / 64, 4), 256, 0, stream>>>(cat_l, gam, bet, mu, var, bnbuf);
    conv3x3_kernel<<<dim3(W / 64, H, 4), 128, 0, stream>>>(bnbuf, w1b, b1, nullptr, ybuf, 0);
    conv3x3_kernel<<<dim3(W / 64, H, 4), 128, 0, stream>>>(ybuf, w2b, b2, bnbuf, fbuf, 1);
    conv1x1_kernel<<<dim3(HWT / 128), 256, 0, stream>>>(fbuf, qwb, qb, Qfb);

    // ---- right branch ----
    bn_kernel<<<dim3(HWT / 64, 4), 256, 0, stream>>>(cat_r, gam, bet, mu, var, bnbuf);
    conv3x3_kernel<<<dim3(W / 64, H, 4), 128, 0, stream>>>(bnbuf, w1b, b1, nullptr, ybuf, 0);
    conv3x3_kernel<<<dim3(W / 64, H, 4), 128, 0, stream>>>(ybuf, w2b, b2, bnbuf, fbuf, 1);
    conv1x1_kernel<<<dim3(HWT / 128), 256, 0, stream>>>(fbuf, swb, sb, Kfb);

    // ---- disparity gathers ----
    gather_kernel<<<dim3((HWT * 64) / 256), 256, 0, stream>>>(
        Qfb, Kfb, x_left, x_right, d_left, d_right, Gq, Gk, Gxl, Gxr);

    // ---- per-window attention + blend ----
    pam_kernel<<<dim3(W / 8, H / 8), 256, 0, stream>>>(
        Qfb, Kfb, Gq, Gk, Gxl, Gxr, x_left, x_right, outL, outR);
}
